// CodeBook_34840774705331
// MI455X (gfx1250) — compile-verified, matrix operand order checked by reading, not run on hardware
//
#include <hip/hip_runtime.h>
#include <hip/hip_bf16.h>

typedef __bf16 bf16_t;
typedef __attribute__((ext_vector_type(16))) __bf16 v16bf;
typedef __attribute__((ext_vector_type(8)))  float  v8f;

// Problem constants
#define BB   32
#define CC   256
#define HWSZ 4096      // 64*64
#define DD   128
#define KK   512
#define NTOK 131072    // BB*HWSZ
#define ZQ_ELEMS 16777216  // BB*DD*HWSZ

// Persistence: 256 WGs x 4 batches x 128 tokens = 131072 tokens
#define NWG     256
#define NBATCH  4

// LDS carve offsets (bytes)
#define OFF_PW   0                      // bf16 [128][256]  = 65536 B
#define OFF_EMB  65536                  // bf16 [512][128]  = 131072 B
#define OFF_ZP   (65536 + 131072)       // bf16 [8][16][128]= 32768 B
#define OFF_BIAS (OFF_ZP + 32768)       // f32  [128]       = 512 B
#define OFF_ESQ  (OFF_BIAS + 512)       // f32  [512]       = 2048 B
#define OFF_IDX  (OFF_ESQ + 2048)       // i32  [128]       = 512 B
#define SMEM_BYTES (OFF_IDX + 512)      // 232448 B  (< 320KB WGP LDS)

static __device__ __forceinline__ v16bf lds_pack16(const bf16_t* p0, const bf16_t* p1) {
  // elements 0..7 <- 8 contiguous bf16 at p0 (ds_load_b128)
  // elements 8..15 <- 8 contiguous bf16 at p1 (ds_load_b128)
  union { v16bf v; float4 f[2]; } u;
  u.f[0] = *reinterpret_cast<const float4*>(p0);
  u.f[1] = *reinterpret_cast<const float4*>(p1);
  return u.v;
}

__global__ void vq_codebook_fused(const float* __restrict__ z,
                                  const float* __restrict__ pw,
                                  const float* __restrict__ pb,
                                  const float* __restrict__ emb,
                                  float* __restrict__ out) {
  extern __shared__ char smem[];
  bf16_t* s_pw   = reinterpret_cast<bf16_t*>(smem + OFF_PW);
  bf16_t* s_emb  = reinterpret_cast<bf16_t*>(smem + OFF_EMB);
  bf16_t* s_zp   = reinterpret_cast<bf16_t*>(smem + OFF_ZP);
  float*  s_bias = reinterpret_cast<float*>(smem + OFF_BIAS);
  float*  s_esq  = reinterpret_cast<float*>(smem + OFF_ESQ);
  int*    s_idx  = reinterpret_cast<int*>(smem + OFF_IDX);

  const int tid  = threadIdx.x;        // 0..255
  const int lane = tid & 31;
  const int wave = tid >> 5;           // 0..7
  const int half = lane >> 4;          // 0 | 1
  const int lcol = lane & 15;

  // ---------------- Phase 0: stage weights into LDS (once per WG) ----------------
  for (int i = tid; i < DD * CC; i += 256) s_pw[i]  = (bf16_t)pw[i];
  for (int i = tid; i < KK * DD; i += 256) s_emb[i] = (bf16_t)emb[i];
  if (tid < DD) s_bias[tid] = pb[tid];
  for (int k = tid; k < KK; k += 256) {
    float s = 0.f;
    const float* er = emb + (size_t)k * DD;
    #pragma unroll 8
    for (int d = 0; d < DD; ++d) { float e = er[d]; s += e * e; }
    s_esq[k] = s;
  }
  __syncthreads();

  // Persistent loop: each WG handles NBATCH batches of 128 tokens.
  for (int it = 0; it < NBATCH; ++it) {
    // Wave owns 16 consecutive tokens (all within one image: 128 | 4096)
    const int batch = blockIdx.x + it * NWG;
    const int tok0  = batch * 128 + wave * 16;
    const int bimg  = tok0 >> 12;        // / HWSZ
    const int hw0   = tok0 & (HWSZ - 1);

    // ---------------- Phase 1: projection zp = z @ pw^T + b ----------------
    // A: 16 tokens x 32 c per step; this lane holds token row = lcol,
    //    K = cbase+{0..7} (elems 0..7) and cbase+16+{0..7} (elems 8..15).
    v8f acc[8] = {};
    const float* zbase = z + (size_t)bimg * CC * HWSZ + (hw0 + lcol);
    for (int ct = 0; ct < 8; ++ct) {
      const int cbase = ct * 32 + half * 8;
      v16bf a;
      #pragma unroll
      for (int e = 0; e < 8; ++e) {
        a[e]     = (bf16_t)zbase[(size_t)(cbase + e) * HWSZ];
        a[e + 8] = (bf16_t)zbase[(size_t)(cbase + 16 + e) * HWSZ];
      }
      #pragma unroll
      for (int dt = 0; dt < 8; ++dt) {
        const bf16_t* brow = s_pw + (dt * 16 + lcol) * CC + cbase;  // B col = d
        v16bf bm = lds_pack16(brow, brow + 16);
        acc[dt] = __builtin_amdgcn_wmma_f32_16x16x32_bf16(
            false, a, false, bm, (short)0, acc[dt], false, false);
      }
    }

    // C/D layout: vgpr r, lane-half h -> token = r + 8h, d-col = lcol.
    // Add bias, stash zp tile to LDS (row-major bf16 [token][d]).
    bf16_t* zp = s_zp + wave * 16 * DD;
    #pragma unroll
    for (int dt = 0; dt < 8; ++dt) {
      const float bias = s_bias[dt * 16 + lcol];
      #pragma unroll
      for (int r = 0; r < 8; ++r) {
        const int t = r + half * 8;
        zp[t * DD + dt * 16 + lcol] = (bf16_t)(acc[dt][r] + bias);
      }
    }
    __syncthreads();

    // ---------------- Phase 2: argmin_k (e_sq[k] - 2 * zp . emb[k]) ----------------
    // Cache A operands (zp) for all 4 K-chunks of D=128.
    v16bf Az[4];
    #pragma unroll
    for (int dc = 0; dc < 4; ++dc) {
      const bf16_t* ar = zp + lcol * DD + dc * 32 + half * 8;
      Az[dc] = lds_pack16(ar, ar + 16);
    }

    float bestv[8]; int besti[8];
    #pragma unroll
    for (int r = 0; r < 8; ++r) { bestv[r] = 3.4e38f; besti[r] = 0; }

    for (int kt = 0; kt < KK / 16; ++kt) {
      v8f dot = {};
      #pragma unroll
      for (int dc = 0; dc < 4; ++dc) {
        const bf16_t* brow = s_emb + (kt * 16 + lcol) * DD + dc * 32 + half * 8;
        v16bf bm = lds_pack16(brow, brow + 16);
        dot = __builtin_amdgcn_wmma_f32_16x16x32_bf16(
            false, Az[dc], false, bm, (short)0, dot, false, false);
      }
      const int   code = kt * 16 + lcol;
      const float esq  = s_esq[code];
      #pragma unroll
      for (int r = 0; r < 8; ++r) {
        const float s = esq - 2.0f * dot[r];
        if (s < bestv[r]) { bestv[r] = s; besti[r] = code; }
      }
    }

    // Cross-lane argmin across the 16 code-columns (xor butterfly, width 16
    // keeps each wave32 half independent; lower index wins ties).
    #pragma unroll
    for (int r = 0; r < 8; ++r) {
      float v = bestv[r]; int ix = besti[r];
      #pragma unroll
      for (int off = 8; off >= 1; off >>= 1) {
        float ov = __shfl_xor(v, off, 16);
        int   oi = __shfl_xor(ix, off, 16);
        if (ov < v || (ov == v && oi < ix)) { v = ov; ix = oi; }
      }
      if (lcol == 0) {
        const int t = r + half * 8;
        s_idx[wave * 16 + t] = ix;
        out[ZQ_ELEMS + tok0 + t] = (float)ix;   // min_indices (value-cast)
      }
    }
    __syncthreads();

    // ---------------- Phase 3: write z_q as [B, D, H, W] (fp32 gather) ----------------
    const int* widx = s_idx + wave * 16;
    float* obase = out + (size_t)bimg * DD * HWSZ + hw0;
    #pragma unroll 4
    for (int i = lane; i < 16 * DD; i += 32) {
      const int d = i >> 4, t = i & 15;
      obase[(size_t)d * HWSZ + t] = emb[(size_t)widx[t] * DD + d];
    }
    __syncthreads();   // LDS (zp/idx) reuse boundary for next batch
  }
}

extern "C" void kernel_launch(void* const* d_in, const int* in_sizes, int n_in,
                              void* d_out, int out_size, void* d_ws, size_t ws_size,
                              hipStream_t stream) {
  const float* z   = (const float*)d_in[0];
  const float* pw  = (const float*)d_in[1];
  const float* pb  = (const float*)d_in[2];
  const float* emb = (const float*)d_in[3];
  float* out = (float*)d_out;

  dim3 grid(NWG), block(256);
  hipLaunchKernelGGL(vq_codebook_fused, grid, block, SMEM_BYTES, stream,
                     z, pw, pb, emb, out);
}
